// GATv2Model_53463752900647
// MI455X (gfx1250) — matrix operand order repaired and checked by reading, not compiled, not run on hardware
//
#include <hip/hip_runtime.h>

#define NN 50000
#define EE 800000
#define ETOT (EE + NN)
#define HC 256
#define NHEADS 4
#define HID 64

typedef __bf16 bf16;
typedef __attribute__((ext_vector_type(16))) __bf16 bf16x16;
typedef __attribute__((ext_vector_type(8)))  __bf16 bf16x8;
typedef __attribute__((ext_vector_type(8)))  float  f32x8;

// ---- monotonic float<->uint key for atomic segment-max ----
__device__ __forceinline__ unsigned fkey(float f) {
    unsigned b = __float_as_uint(f);
    return (b & 0x80000000u) ? ~b : (b | 0x80000000u);
}
__device__ __forceinline__ float funkey(unsigned k) {
    unsigned b = (k & 0x80000000u) ? (k & 0x7fffffffu) : ~k;
    return __uint_as_float(b);
}

// ---- f32 -> bf16 convert ----
__global__ void k_cvt_bf16(const float* __restrict__ in, bf16* __restrict__ out, int n) {
    int i = blockIdx.x * blockDim.x + threadIdx.x;
    if (i < n) out[i] = (bf16)in[i];
}

// ---- W[k][n] (f32, 256x256) -> Wt[n][k] (bf16) so B-fragment K-values are contiguous ----
__global__ void k_prep_wt(const float* __restrict__ W, bf16* __restrict__ Wt) {
    int i = blockIdx.x * blockDim.x + threadIdx.x;
    if (i < HC * HC) {
        int k = i >> 8, n = i & 255;
        Wt[n * HC + k] = (bf16)W[i];
    }
}

// ---- C[M x 256] = A[M x 256](bf16) @ W (via Wt[n][k] bf16), f32 accumulate ----
// One wave computes a 16(M) x 64(N) strip: 8 K-steps x 4 N-tiles = 32 WMMA.
__global__ void k_gemm_bf16(const bf16* __restrict__ A, const bf16* __restrict__ Wt,
                            float* __restrict__ C) {
    int gt   = blockIdx.x * blockDim.x + threadIdx.x;
    int wid  = gt >> 5;
    int lane = gt & 31;
    if (wid >= (NN / 16) * 4) return;
    int mtile = wid >> 2, ng = wid & 3;
    int m0 = mtile * 16, n0 = ng * 64;
    int hf = lane >> 4, l15 = lane & 15;

    f32x8 acc[4];
#pragma unroll
    for (int t = 0; t < 4; ++t)
#pragma unroll
        for (int r = 0; r < 8; ++r) acc[t][r] = 0.0f;

    const bf16* arow = A + (size_t)(m0 + l15) * HC;
#pragma unroll
    for (int kk = 0; kk < HC; kk += 32) {
        // A fragment (16x32 bf16): lanes 0-15 hold K {kk..kk+7, kk+16..kk+23},
        // lanes 16-31 hold K {kk+8..kk+15, kk+24..kk+31}
        int kb = kk + hf * 8;
        bf16x8 alo = *(const bf16x8*)(arow + kb);
        bf16x8 ahi = *(const bf16x8*)(arow + kb + 16);
        bf16x16 afrag = __builtin_shufflevector(alo, ahi,
            0, 1, 2, 3, 4, 5, 6, 7, 8, 9, 10, 11, 12, 13, 14, 15);
#pragma unroll
        for (int t = 0; t < 4; ++t) {
            // B fragment (32x16 bf16): lane holds column n0+t*16+l15,
            // K {kk+hf*8..+7, kk+16+hf*8..+7}, matching the A fragment.
            const bf16* wrow = Wt + (size_t)(n0 + t * 16 + l15) * HC + kk + hf * 8;
            bf16x8 b0 = *(const bf16x8*)wrow;
            bf16x8 b1 = *(const bf16x8*)(wrow + 16);
            bf16x16 bfrag = __builtin_shufflevector(b0, b1,
                0, 1, 2, 3, 4, 5, 6, 7, 8, 9, 10, 11, 12, 13, 14, 15);
            acc[t] = __builtin_amdgcn_wmma_f32_16x16x32_bf16(
                false, afrag, false, bfrag, (short)0, acc[t], false, false);
        }
    }
    // D layout: VGPR r -> row m0 + r + hf*8, col = n0 + t*16 + l15
#pragma unroll
    for (int t = 0; t < 4; ++t)
#pragma unroll
        for (int r = 0; r < 8; ++r)
            C[(size_t)(m0 + r + hf * 8) * HC + n0 + t * 16 + l15] = acc[t][r];
}

// ---- per (edge, head): attention logit + atomic segment max; one wave each ----
__global__ void k_edge_logits(const long long* __restrict__ ei,
                              const float* __restrict__ xl, const float* __restrict__ xr,
                              const float* __restrict__ att,
                              float* __restrict__ alpha, unsigned* __restrict__ amax) {
    int gt = blockIdx.x * blockDim.x + threadIdx.x;
    int wid = gt >> 5, lane = gt & 31;
    if (wid >= ETOT * NHEADS) return;
    int e = wid >> 2, h = wid & 3;
    int src, dst;
    if (e < EE) { src = (int)ei[e]; dst = (int)ei[EE + e]; }
    else        { src = dst = e - EE; }
    const float* pl = xl + (size_t)src * HC + h * HID;
    const float* pr = xr + (size_t)dst * HC + h * HID;
    const float* pa = att + h * HID;
    float s = 0.0f;
#pragma unroll
    for (int j = 0; j < 2; ++j) {
        int c = lane + j * 32;
        float v = pl[c] + pr[c];
        v = v > 0.0f ? v : 0.2f * v;   // LeakyReLU(0.2)
        s += v * pa[c];
    }
#pragma unroll
    for (int m = 16; m > 0; m >>= 1) s += __shfl_xor(s, m, 32);
    if (lane == 0) {
        alpha[wid] = s;
        atomicMax(&amax[dst * NHEADS + h], fkey(s));
    }
}

// ---- exp(alpha - max) and denominator accumulation ----
__global__ void k_edge_norm(const long long* __restrict__ ei, float* __restrict__ alpha,
                            const unsigned* __restrict__ amax, float* __restrict__ denom) {
    int i = blockIdx.x * blockDim.x + threadIdx.x;
    if (i >= ETOT * NHEADS) return;
    int e = i >> 2, h = i & 3;
    int dst = (e < EE) ? (int)ei[EE + e] : (e - EE);
    float v = __expf(alpha[i] - funkey(amax[dst * NHEADS + h]));
    alpha[i] = v;
    atomicAdd(&denom[dst * NHEADS + h], v);
}

// ---- weighted scatter-add: out[dst] += alpha/denom * xl[src]; one wave per (e,h) ----
__global__ void k_edge_scatter(const long long* __restrict__ ei, const float* __restrict__ xl,
                               const float* __restrict__ alpha, const float* __restrict__ denom,
                               float* __restrict__ out) {
    int gt = blockIdx.x * blockDim.x + threadIdx.x;
    int wid = gt >> 5, lane = gt & 31;
    if (wid >= ETOT * NHEADS) return;
    int e = wid >> 2, h = wid & 3;
    int src, dst;
    if (e < EE) { src = (int)ei[e]; dst = (int)ei[EE + e]; }
    else        { src = dst = e - EE; }
    float w = alpha[wid] / (denom[dst * NHEADS + h] + 1e-16f);
    const float* pl = xl + (size_t)src * HC + h * HID;
    float* po = out + (size_t)dst * HC + h * HID;
    atomicAdd(&po[lane],      w * pl[lane]);
    atomicAdd(&po[lane + 32], w * pl[lane + 32]);
}

// ---- bias + ELU in place, optional bf16 re-encode for next layer's GEMM ----
__global__ void k_finish(float* __restrict__ h, const float* __restrict__ bias,
                         bf16* __restrict__ xb, int writeBf) {
    int i = blockIdx.x * blockDim.x + threadIdx.x;
    if (i >= NN * HC) return;
    float v = h[i] + bias[i & 255];
    v = v > 0.0f ? v : (__expf(v) - 1.0f);  // ELU
    h[i] = v;
    if (writeBf) xb[i] = (bf16)v;
}

// ---- layer-2 projections: [N x 256] @ [256 x 2], both Wl2 and Wr2 ----
__global__ void k_proj2(const float* __restrict__ h, const float* __restrict__ Wl,
                        const float* __restrict__ Wr,
                        float* __restrict__ xl2, float* __restrict__ xr2) {
    int n = blockIdx.x * blockDim.x + threadIdx.x;
    if (n >= NN) return;
    const float* p = h + (size_t)n * HC;
    float l0 = 0, l1 = 0, r0 = 0, r1 = 0;
    for (int k = 0; k < HC; ++k) {
        float v = p[k];
        l0 += v * Wl[k * 2]; l1 += v * Wl[k * 2 + 1];
        r0 += v * Wr[k * 2]; r1 += v * Wr[k * 2 + 1];
    }
    xl2[n * 2] = l0; xl2[n * 2 + 1] = l1;
    xr2[n * 2] = r0; xr2[n * 2 + 1] = r1;
}

__global__ void k_edge2_logit(const long long* __restrict__ ei, const float* __restrict__ xl2,
                              const float* __restrict__ xr2, const float* __restrict__ att2,
                              float* __restrict__ alpha2, unsigned* __restrict__ amax2) {
    int e = blockIdx.x * blockDim.x + threadIdx.x;
    if (e >= ETOT) return;
    int src, dst;
    if (e < EE) { src = (int)ei[e]; dst = (int)ei[EE + e]; }
    else        { src = dst = e - EE; }
    float e0 = xl2[src * 2]     + xr2[dst * 2];     e0 = e0 > 0.f ? e0 : 0.2f * e0;
    float e1 = xl2[src * 2 + 1] + xr2[dst * 2 + 1]; e1 = e1 > 0.f ? e1 : 0.2f * e1;
    float a = att2[0] * e0 + att2[1] * e1;
    alpha2[e] = a;
    atomicMax(&amax2[dst], fkey(a));
}

__global__ void k_edge2_norm(const long long* __restrict__ ei, float* __restrict__ alpha2,
                             const unsigned* __restrict__ amax2, float* __restrict__ denom2) {
    int e = blockIdx.x * blockDim.x + threadIdx.x;
    if (e >= ETOT) return;
    int dst = (e < EE) ? (int)ei[EE + e] : (e - EE);
    float v = __expf(alpha2[e] - funkey(amax2[dst]));
    alpha2[e] = v;
    atomicAdd(&denom2[dst], v);
}

__global__ void k_edge2_scatter(const long long* __restrict__ ei, const float* __restrict__ xl2,
                                const float* __restrict__ alpha2, const float* __restrict__ denom2,
                                float* __restrict__ out) {
    int e = blockIdx.x * blockDim.x + threadIdx.x;
    if (e >= ETOT) return;
    int src, dst;
    if (e < EE) { src = (int)ei[e]; dst = (int)ei[EE + e]; }
    else        { src = dst = e - EE; }
    float w = alpha2[e] / (denom2[dst] + 1e-16f);
    atomicAdd(&out[dst * 2],     w * xl2[src * 2]);
    atomicAdd(&out[dst * 2 + 1], w * xl2[src * 2 + 1]);
}

__global__ void k_bias2(float* __restrict__ out, const float* __restrict__ b2) {
    int n = blockIdx.x * blockDim.x + threadIdx.x;
    if (n >= NN) return;
    out[n * 2]     += b2[0];
    out[n * 2 + 1] += b2[1];
}

extern "C" void kernel_launch(void* const* d_in, const int* in_sizes, int n_in,
                              void* d_out, int out_size, void* d_ws, size_t ws_size,
                              hipStream_t stream) {
    (void)in_sizes; (void)n_in; (void)ws_size;
    const float*     x    = (const float*)d_in[0];
    const long long* ei   = (const long long*)d_in[1];
    const float*     Wl0  = (const float*)d_in[2];
    const float*     Wr0  = (const float*)d_in[3];
    const float*     att0 = (const float*)d_in[4];
    const float*     b0   = (const float*)d_in[5];
    const float*     Wl1  = (const float*)d_in[6];
    const float*     Wr1  = (const float*)d_in[7];
    const float*     att1 = (const float*)d_in[8];
    const float*     b1   = (const float*)d_in[9];
    const float*     Wl2  = (const float*)d_in[10];
    const float*     Wr2  = (const float*)d_in[11];
    const float*     att2 = (const float*)d_in[12];
    const float*     b2   = (const float*)d_in[13];
    float* out = (float*)d_out;

    char* ws = (char*)d_ws; size_t off = 0;
    auto alloc = [&](size_t bytes) -> char* {
        char* p = ws + off;
        off = (off + bytes + 255) & ~(size_t)255;
        return p;
    };
    bf16*     bfX    = (bf16*)alloc((size_t)NN * HC * 2);
    float*    xl     = (float*)alloc((size_t)NN * HC * 4);
    float*    xr     = (float*)alloc((size_t)NN * HC * 4);
    float*    hh     = (float*)alloc((size_t)NN * HC * 4);
    float*    alpha  = (float*)alloc((size_t)ETOT * NHEADS * 4);
    unsigned* amax   = (unsigned*)alloc((size_t)NN * NHEADS * 4);
    float*    denom  = (float*)alloc((size_t)NN * NHEADS * 4);
    bf16* Wt[4];
    for (int i = 0; i < 4; ++i) Wt[i] = (bf16*)alloc((size_t)HC * HC * 2);
    float*    xl2    = (float*)alloc((size_t)NN * 2 * 4);
    float*    xr2    = (float*)alloc((size_t)NN * 2 * 4);
    float*    alpha2 = (float*)alloc((size_t)ETOT * 4);
    unsigned* amax2  = (unsigned*)alloc((size_t)NN * 4);
    float*    denom2 = (float*)alloc((size_t)NN * 4);

    const int B = 256;
    const int gemmBlocks = ((NN / 16) * 4 + 7) / 8;      // 8 waves / block
    const int ewBlocks   = (ETOT * NHEADS) / 8;          // wave per (edge, head)
    const int etBlocks   = (ETOT * NHEADS + B - 1) / B;  // thread per (edge, head)
    const int eBlocks    = (ETOT + B - 1) / B;
    const int nkBlocks   = (NN * HC + B - 1) / B;

    // prep: bf16 weights (transposed) + bf16 x
    const float* Wsrc[4] = {Wl0, Wr0, Wl1, Wr1};
    for (int i = 0; i < 4; ++i)
        k_prep_wt<<<(HC * HC + B - 1) / B, B, 0, stream>>>(Wsrc[i], Wt[i]);
    k_cvt_bf16<<<nkBlocks, B, 0, stream>>>(x, bfX, NN * HC);

    // ---- layer 0 ----
    k_gemm_bf16<<<gemmBlocks, B, 0, stream>>>(bfX, Wt[0], xl);
    k_gemm_bf16<<<gemmBlocks, B, 0, stream>>>(bfX, Wt[1], xr);
    hipMemsetAsync(hh,    0, (size_t)NN * HC * 4,     stream);
    hipMemsetAsync(amax,  0, (size_t)NN * NHEADS * 4, stream);
    hipMemsetAsync(denom, 0, (size_t)NN * NHEADS * 4, stream);
    k_edge_logits <<<ewBlocks, B, 0, stream>>>(ei, xl, xr, att0, alpha, amax);
    k_edge_norm   <<<etBlocks, B, 0, stream>>>(ei, alpha, amax, denom);
    k_edge_scatter<<<ewBlocks, B, 0, stream>>>(ei, xl, alpha, denom, hh);
    k_finish      <<<nkBlocks, B, 0, stream>>>(hh, b0, bfX, 1);

    // ---- layer 1 ----
    k_gemm_bf16<<<gemmBlocks, B, 0, stream>>>(bfX, Wt[2], xl);
    k_gemm_bf16<<<gemmBlocks, B, 0, stream>>>(bfX, Wt[3], xr);
    hipMemsetAsync(hh,    0, (size_t)NN * HC * 4,     stream);
    hipMemsetAsync(amax,  0, (size_t)NN * NHEADS * 4, stream);
    hipMemsetAsync(denom, 0, (size_t)NN * NHEADS * 4, stream);
    k_edge_logits <<<ewBlocks, B, 0, stream>>>(ei, xl, xr, att1, alpha, amax);
    k_edge_norm   <<<etBlocks, B, 0, stream>>>(ei, alpha, amax, denom);
    k_edge_scatter<<<ewBlocks, B, 0, stream>>>(ei, xl, alpha, denom, hh);
    k_finish      <<<nkBlocks, B, 0, stream>>>(hh, b1, bfX, 0);

    // ---- layer 2 (heads=1, C=2, concat=False) ----
    k_proj2<<<(NN + B - 1) / B, B, 0, stream>>>(hh, Wl2, Wr2, xl2, xr2);
    hipMemsetAsync(amax2,  0, (size_t)NN * 4, stream);
    hipMemsetAsync(denom2, 0, (size_t)NN * 4, stream);
    hipMemsetAsync(d_out,  0, (size_t)out_size * 4, stream);
    k_edge2_logit  <<<eBlocks, B, 0, stream>>>(ei, xl2, xr2, att2, alpha2, amax2);
    k_edge2_norm   <<<eBlocks, B, 0, stream>>>(ei, alpha2, amax2, denom2);
    k_edge2_scatter<<<eBlocks, B, 0, stream>>>(ei, xl2, alpha2, denom2, out);
    k_bias2        <<<(NN + B - 1) / B, B, 0, stream>>>(out, b2);
}
